// GNNClassifier_88218628260533
// MI455X (gfx1250) — matrix operand order repaired and checked by reading, not compiled, not run on hardware
//
#include <hip/hip_runtime.h>
#include <hip/hip_bf16.h>

typedef _Float16 half_t;
typedef __attribute__((ext_vector_type(16))) _Float16 v16h;
typedef __attribute__((ext_vector_type(8)))  float    v8f;

#define Bsz 32
#define Nn  1024
#define Kn  16
#define Hh  64

// 16 halfs viewed as two float4 (two 8-half chunks: K..K+7 and K+16..K+23)
union AF {
    v16h   v;
    float4 f4[2];
};

// ---------------------------------------------------------------------------
// Repack a (K x 64) f32 row-major weight into the per-lane WMMA B layout (f16).
// Layout: tile (t,nt), plane p in {0,1}, lane l, j in 0..7:
//   idx = ((t*4+nt)*512) + p*256 + l*8 + j
//   k   = t*32 + (l>>4)*8 + p*16 + j ; n = nt*16 + (l&15)
// ---------------------------------------------------------------------------
__global__ void repack_kernel(const float* __restrict__ w, half_t* __restrict__ out,
                              int K, int ksteps) {
    int idx = blockIdx.x * blockDim.x + threadIdx.x;
    int total = ksteps * 4 * 512;
    if (idx >= total) return;
    int j    = idx & 7;
    int lane = (idx >> 3) & 31;
    int p    = (idx >> 8) & 1;
    int tile = idx >> 9;          // t*4 + nt
    int nt   = tile & 3;
    int t    = tile >> 2;
    int hs   = lane >> 4;
    int e    = lane & 15;
    int k = t * 32 + hs * 8 + p * 16 + j;
    int n = nt * 16 + e;
    float v = (k < K) ? w[k * 64 + n] : 0.0f;
    out[idx] = (half_t)v;
}

// ---------------------------------------------------------------------------
// Embed: x0 = relu(LN(nf @ W + b)) -> f16. One wave (32 lanes) per node,
// each lane owns 2 of the 64 channels; LN via wave shuffle reduction.
// ---------------------------------------------------------------------------
__global__ __launch_bounds__(256)
void embed_kernel(const float* __restrict__ nf, const float* __restrict__ w,
                  const float* __restrict__ bias, const float* __restrict__ lg,
                  const float* __restrict__ lb, half_t* __restrict__ x0) {
    int tid  = threadIdx.x;
    int wave = tid >> 5, lane = tid & 31;
    int node = blockIdx.x * 8 + wave;           // 4096 blocks * 8 waves = 32768
    const float* row = nf + node * 6;
    float f[6];
#pragma unroll
    for (int k = 0; k < 6; ++k) f[k] = row[k];

    float y[2];
#pragma unroll
    for (int i = 0; i < 2; ++i) {
        int c = lane * 2 + i;
        float s = bias[c];
#pragma unroll
        for (int k = 0; k < 6; ++k) s += f[k] * w[k * 64 + c];
        y[i] = s;
    }
    float s = y[0] + y[1];
#pragma unroll
    for (int off = 16; off > 0; off >>= 1) s += __shfl_xor(s, off);
    float mean = s * (1.0f / 64.0f);
    float q = (y[0] - mean) * (y[0] - mean) + (y[1] - mean) * (y[1] - mean);
#pragma unroll
    for (int off = 16; off > 0; off >>= 1) q += __shfl_xor(q, off);
    float rstd = rsqrtf(q * (1.0f / 64.0f) + 1e-5f);
#pragma unroll
    for (int i = 0; i < 2; ++i) {
        int c = lane * 2 + i;
        float v = (y[i] - mean) * rstd * lg[c] + lb[c];
        v = fmaxf(v, 0.0f);
        x0[node * 64 + c] = (half_t)v;
    }
}

// ---------------------------------------------------------------------------
// Message-passing layer. 256 threads = 8 waves; each wave owns 16 nodes.
// Edge phase: per node a 16(edges) x 160(K,padded) x 64 GEMM via 5x4 WMMAs,
// A built in registers straight from global f16 x (self / neighbor / rel_pos
// regions are 64-aligned so 8-half chunks never straddle).
// Node phase: [x | msg] 16x128 LDS tile -> 4x4 WMMAs vs nw, then LN+ReLU+mask.
// ---------------------------------------------------------------------------
__global__ __launch_bounds__(256)
void mp_kernel(const half_t* __restrict__ xin, half_t* __restrict__ xout,
               const int* __restrict__ ei, const float* __restrict__ nf,
               const float* __restrict__ maskp,
               const half_t* __restrict__ ew_pk, const float* __restrict__ eb,
               const half_t* __restrict__ nw_pk, const float* __restrict__ nb,
               const float* __restrict__ lg, const float* __restrict__ lb) {
    __shared__ __attribute__((aligned(16))) half_t s_ew[5 * 4 * 512];   // 20 KB
    __shared__ __attribute__((aligned(16))) half_t s_nw[4 * 4 * 512];   // 16 KB
    __shared__ __attribute__((aligned(16))) half_t s_tile[8][16 * 128]; // 32 KB

    int tid = threadIdx.x;
    // cooperative preload of packed weights into LDS (float4 = 8 halfs)
    for (int i = tid; i < (5 * 4 * 512) / 8; i += 256)
        ((float4*)s_ew)[i] = ((const float4*)ew_pk)[i];
    for (int i = tid; i < (4 * 4 * 512) / 8; i += 256)
        ((float4*)s_nw)[i] = ((const float4*)nw_pk)[i];
    __syncthreads();

    int wave = tid >> 5, lane = tid & 31;
    int gw    = blockIdx.x * 8 + wave;   // 256 blocks * 8 waves = 2048 waves
    int node0 = gw * 16;
    int b  = node0 >> 10;
    int n0 = node0 & (Nn - 1);
    int e  = lane & 15;
    int hs = lane >> 4;

    half_t* tile = s_tile[wave];
    v8f acc[4];

    // -------- edge phase: 16 nodes, one at a time --------
    for (int m = 0; m < 16; ++m) {
        int n = n0 + m;
        int selfRow = ((b << 10) + n) * 64;                 // half offset
        int nbr  = ei[(((b << 10) + n) << 4) + e];
        int nbrRow = ((b << 10) + nbr) * 64;
        float px = nf[((b << 10) + n) * 6 + 4];
        float py = nf[((b << 10) + n) * 6 + 5];
        float rx = nf[((b << 10) + nbr) * 6 + 4] - px;
        float ry = nf[((b << 10) + nbr) * 6 + 5] - py;

#pragma unroll
        for (int nt = 0; nt < 4; ++nt) {
            float bv = eb[nt * 16 + e];
#pragma unroll
            for (int r = 0; r < 8; ++r) acc[nt][r] = bv;
        }

#pragma unroll
        for (int t = 0; t < 5; ++t) {
            AF a;
            if (t < 2) {
                const float4* p = (const float4*)(xin + selfRow + t * 32 + hs * 8);
                a.f4[0] = p[0];
                a.f4[1] = p[2];   // +16 halfs
            } else if (t < 4) {
                const float4* p = (const float4*)(xin + nbrRow + (t - 2) * 32 + hs * 8);
                a.f4[0] = p[0];
                a.f4[1] = p[2];
            } else {
                a.f4[0] = make_float4(0.f, 0.f, 0.f, 0.f);
                a.f4[1] = make_float4(0.f, 0.f, 0.f, 0.f);
                if (hs == 0) { a.v[0] = (half_t)rx; a.v[1] = (half_t)ry; }
            }
#pragma unroll
            for (int nt = 0; nt < 4; ++nt) {
                AF bm;
                const float4* wp = (const float4*)(s_ew + (t * 4 + nt) * 512);
                bm.f4[0] = wp[lane];
                bm.f4[1] = wp[32 + lane];
                acc[nt] = __builtin_amdgcn_wmma_f32_16x16x32_f16(
                    false, a.v, false, bm.v, (short)0, acc[nt], false, false);
            }
        }

        // relu + mean over the 16 edges (rows)
#pragma unroll
        for (int nt = 0; nt < 4; ++nt) {
            float s = 0.f;
#pragma unroll
            for (int r = 0; r < 8; ++r) {
                float v = acc[nt][r];
                s += fmaxf(v, 0.0f);
            }
            s += __shfl_xor(s, 16);
            s *= (1.0f / 16.0f);
            if (hs == 0) tile[m * 128 + 64 + nt * 16 + e] = (half_t)s;
        }
        // copy self x row (64 halfs) into tile: each lane moves 2 halfs
        ((float*)(tile + m * 128))[lane] = ((const float*)(xin + selfRow))[lane];
    }

    // -------- node phase: 16x128 @ 128x64 --------
    v8f h[4];
#pragma unroll
    for (int nt = 0; nt < 4; ++nt) {
        float bv = nb[nt * 16 + e];
#pragma unroll
        for (int r = 0; r < 8; ++r) h[nt][r] = bv;
    }
#pragma unroll
    for (int t = 0; t < 4; ++t) {
        AF a;
        const float4* p = (const float4*)(tile + e * 128 + t * 32 + hs * 8);
        a.f4[0] = p[0];
        a.f4[1] = p[2];
#pragma unroll
        for (int nt = 0; nt < 4; ++nt) {
            AF bm;
            const float4* wp = (const float4*)(s_nw + (t * 4 + nt) * 512);
            bm.f4[0] = wp[lane];
            bm.f4[1] = wp[32 + lane];
            h[nt] = __builtin_amdgcn_wmma_f32_16x16x32_f16(
                false, a.v, false, bm.v, (short)0, h[nt], false, false);
        }
    }

    // LayerNorm per row (row = node, cols striped over 16-lane halves x 4 nt)
    float mean[8], rstd[8];
#pragma unroll
    for (int r = 0; r < 8; ++r) {
        float s = 0.f, q = 0.f;
#pragma unroll
        for (int nt = 0; nt < 4; ++nt) {
            float v = h[nt][r];
            s += v;
            q += v * v;
        }
#pragma unroll
        for (int off = 1; off <= 8; off <<= 1) {
            s += __shfl_xor(s, off);
            q += __shfl_xor(q, off);
        }
        float mu = s * (1.0f / 64.0f);
        float var = q * (1.0f / 64.0f) - mu * mu;
        mean[r] = mu;
        rstd[r] = rsqrtf(var + 1e-5f);
    }
    float gm[4], bt[4];
#pragma unroll
    for (int nt = 0; nt < 4; ++nt) {
        gm[nt] = lg[nt * 16 + e];
        bt[nt] = lb[nt * 16 + e];
    }
#pragma unroll
    for (int r = 0; r < 8; ++r) {
        int mrow = r + 8 * hs;
        float mk = maskp[(b << 10) + n0 + mrow];
        int ob = ((b << 10) + n0 + mrow) * 64;
#pragma unroll
        for (int nt = 0; nt < 4; ++nt) {
            float v = (h[nt][r] - mean[r]) * rstd[r] * gm[nt] + bt[nt];
            v = fmaxf(v, 0.0f) * mk;
            xout[ob + nt * 16 + e] = (half_t)v;
        }
    }
}

// ---------------------------------------------------------------------------
// Masked mean pooling over nodes: pooled[b][c], f32. One block per batch.
// ---------------------------------------------------------------------------
__global__ __launch_bounds__(256)
void pool_kernel(const half_t* __restrict__ x, const float* __restrict__ maskp,
                 float* __restrict__ pooled) {
    int b = blockIdx.x, tid = threadIdx.x;
    int c = tid & 63, chunk = tid >> 6;   // 4 chunks of 256 nodes
    float s = 0.f;
    for (int n = chunk * 256; n < chunk * 256 + 256; ++n)
        s += (float)x[((b << 10) + n) * 64 + c] * maskp[(b << 10) + n];
    __shared__ float red[256];
    __shared__ float redm[64];
    red[tid] = s;
    if (tid < 64) {
        float ms = 0.f;
        for (int i = 0; i < 16; ++i) ms += maskp[(b << 10) + tid * 16 + i];
        redm[tid] = ms;
    }
    __syncthreads();
    if (tid == 0) {
        float m = 0.f;
        for (int i = 0; i < 64; ++i) m += redm[i];
        redm[0] = fmaxf(m, 1.0f);
    }
    __syncthreads();
    if (chunk == 0) {
        float tot = red[c] + red[64 + c] + red[128 + c] + red[192 + c];
        pooled[b * 64 + c] = tot / redm[0];
    }
}

// ---------------------------------------------------------------------------
// Readout + reco branch + head. One block (128 threads) per batch row.
// ---------------------------------------------------------------------------
__global__ __launch_bounds__(128)
void tail_kernel(const float* __restrict__ pooled, const float* __restrict__ reco,
                 const float* __restrict__ ro_w, const float* __restrict__ ro_b,
                 const float* __restrict__ rc_w1, const float* __restrict__ rc_b1,
                 const float* __restrict__ rc_w2, const float* __restrict__ rc_b2,
                 const float* __restrict__ hd_w1, const float* __restrict__ hd_b1,
                 const float* __restrict__ hd_w2, const float* __restrict__ hd_b2,
                 float* __restrict__ out) {
    int b = blockIdx.x, t = threadIdx.x;
    __shared__ float g[64], r1[64], r2[64], hbuf[128];
    if (t < 64) {
        float s = ro_b[t];
        for (int k = 0; k < 64; ++k) s += pooled[b * 64 + k] * ro_w[k * 64 + t];
        g[t] = fmaxf(s, 0.0f);
        float s1 = rc_b1[t];
        for (int k = 0; k < 6; ++k) s1 += reco[b * 6 + k] * rc_w1[k * 64 + t];
        r1[t] = fmaxf(s1, 0.0f);
    }
    __syncthreads();
    if (t < 64) {
        float s = rc_b2[t];
        for (int k = 0; k < 64; ++k) s += r1[k] * rc_w2[k * 64 + t];
        r2[t] = fmaxf(s, 0.0f);
    }
    __syncthreads();
    float s = hd_b1[t];
    for (int k = 0; k < 64; ++k) s += g[k] * hd_w1[k * 128 + t];
    for (int k = 0; k < 64; ++k) s += r2[k] * hd_w1[(64 + k) * 128 + t];
    hbuf[t] = fmaxf(s, 0.0f);
    __syncthreads();
    if (t < 5) {
        float o = hd_b2[t];
        for (int k = 0; k < 128; ++k) o += hbuf[k] * hd_w2[k * 5 + t];
        out[b * 5 + t] = o;
    }
}

// ---------------------------------------------------------------------------
extern "C" void kernel_launch(void* const* d_in, const int* in_sizes, int n_in,
                              void* d_out, int out_size, void* d_ws, size_t ws_size,
                              hipStream_t stream) {
    auto F = [&](int i) { return (const float*)d_in[i]; };
    const float* nf    = F(0);
    const int*   ei    = (const int*)d_in[1];
    const float* maskp = F(2);
    const float* reco  = F(3);
    const float* embed_w  = F(4);
    const float* embed_b  = F(5);
    const float* embed_lg = F(6);
    const float* embed_lb = F(7);
    // mp layer l params at 8 + l*6: ew, eb, nw, nb, lg, lb
    const float* ro_w  = F(26);
    const float* ro_b  = F(27);
    const float* rc_w1 = F(28);
    const float* rc_b1 = F(29);
    const float* rc_w2 = F(30);
    const float* rc_b2 = F(31);
    const float* hd_w1 = F(32);
    const float* hd_b1 = F(33);
    const float* hd_w2 = F(34);
    const float* hd_b2 = F(35);

    // workspace layout
    const size_t XELEMS = (size_t)Bsz * Nn * Hh;       // 2,097,152 halfs
    char* ws = (char*)d_ws;
    half_t* x0   = (half_t*)ws;
    half_t* x1   = (half_t*)(ws + XELEMS * sizeof(half_t));
    half_t* ewpk = (half_t*)(ws + 2 * XELEMS * sizeof(half_t));   // 3 * 10240 halfs
    half_t* nwpk = ewpk + 3 * 10240;                              // 3 *  8192 halfs
    float*  pooled = (float*)(nwpk + 3 * 8192);                   // 32*64 f32

    // repack weights (f32 -> f16 WMMA B layout)
    for (int l = 0; l < 3; ++l) {
        const float* ew = F(8 + l * 6 + 0);
        const float* nw = F(8 + l * 6 + 2);
        repack_kernel<<<(10240 + 255) / 256, 256, 0, stream>>>(ew, ewpk + l * 10240, 130, 5);
        repack_kernel<<<(8192 + 255) / 256, 256, 0, stream>>>(nw, nwpk + l * 8192, 128, 4);
    }

    // embed
    embed_kernel<<<4096, 256, 0, stream>>>(nf, embed_w, embed_b, embed_lg, embed_lb, x0);

    // 3 message-passing layers (ping-pong x buffers)
    half_t* xc = x0;
    half_t* xn = x1;
    for (int l = 0; l < 3; ++l) {
        const float* eb = F(8 + l * 6 + 1);
        const float* nb = F(8 + l * 6 + 3);
        const float* lg = F(8 + l * 6 + 4);
        const float* lb = F(8 + l * 6 + 5);
        mp_kernel<<<256, 256, 0, stream>>>(xc, xn, ei, nf, maskp,
                                           ewpk + l * 10240, eb,
                                           nwpk + l * 8192, nb, lg, lb);
        half_t* tmp = xc; xc = xn; xn = tmp;
    }

    // pooling + head
    pool_kernel<<<Bsz, 256, 0, stream>>>(xc, maskp, pooled);
    tail_kernel<<<Bsz, 128, 0, stream>>>(pooled, reco, ro_w, ro_b, rc_w1, rc_b1,
                                         rc_w2, rc_b2, hd_w1, hd_b1, hd_w2, hd_b2,
                                         (float*)d_out);
}
